// CrossAttentionLayer_two_level_rep_54735063220675
// MI455X (gfx1250) — matrix operand order repaired and checked by reading, not compiled, not run on hardware
//
#include <hip/hip_runtime.h>
#include <hip/hip_bf16.h>

typedef __attribute__((ext_vector_type(16))) _Float16 v16h;
typedef __attribute__((ext_vector_type(8)))  float    v8f;
typedef __attribute__((ext_vector_type(2)))  int      v2i;

#define QDIM   128
#define HDIM   16
#define LROWS  54            // 6*9 spatial positions per batch
#define PAIRB  2             // batches per block-pair
#define PROWS  (LROWS*PAIRB) // 108 rows per pair
#define MTILES 7             // ceil(108/16) -> 112 padded rows
#define MPAD   (MTILES*16)
#define SXL    136           // padded LDS row stride (halves)
#define SRL    136
#define MATH   16384         // halves per swizzled weight matrix
#define PAIR_FLOATS (PROWS*QDIM)   // 13824 floats = 55296 B
#define ACHUNK (PAIR_FLOATS/2/256) // 27 b64 copies per thread per stage
#define PPB    4             // batch-pairs per persistent block

// ---- dynamic LDS partition (bytes, all 16B aligned) ------------------------
#define OFF_XRAW0 0
#define OFF_XRAW1 55296
#define OFF_SX    110592     // 112 x 136 halves  = 30464 B
#define OFF_SK    141056     // 108 x 128 halves  = 27648 B
#define OFF_SV    168704     // 108 x 128 halves  = 27648 B
#define OFF_ATT   196352     // 16 x 128 f32      =  8192 B
#define OFF_XRES  204544     // 16 x 128 f32      =  8192 B
#define OFF_SA    212736     // 16 x 136 halves   =  4352 B
#define OFF_SB2   217088     // 16 x 136 halves   =  4352 B
#define OFF_STAT  221440     // mean[16]+rstd[16] =   128 B
#define SMEM_BYTES 221568

// ---- gfx1250 async global->LDS path (guarded) ------------------------------
#if defined(__has_builtin)
#  if __has_builtin(__builtin_amdgcn_global_load_async_to_lds_b64)
#    define HAVE_ASYNC_LDS 1
#  endif
#endif
#ifndef HAVE_ASYNC_LDS
#  define HAVE_ASYNC_LDS 0
#endif

#if HAVE_ASYNC_LDS
typedef __attribute__((address_space(1))) v2i gv2i;
typedef __attribute__((address_space(3))) v2i lv2i;
#endif

__device__ __forceinline__ void stage_pair(const float* __restrict__ gsrc,
                                           float* lraw, int tid) {
#if HAVE_ASYNC_LDS
  gv2i* g = (gv2i*)gsrc;   // addrspacecast generic->global, drop const
  lv2i* l = (lv2i*)lraw;   // addrspacecast generic->local
  for (int i = tid; i < PAIR_FLOATS / 2; i += 256)
    __builtin_amdgcn_global_load_async_to_lds_b64(g + i, l + i, 0, 0);
#else
  const float2* s = (const float2*)gsrc;
  float2*       d = (float2*)lraw;
  for (int i = tid; i < PAIR_FLOATS / 2; i += 256) d[i] = s[i];
#endif
}

__device__ __forceinline__ void wait_stage(int more_in_flight) {
#if HAVE_ASYNC_LDS
#  if __has_builtin(__builtin_amdgcn_s_wait_asynccnt)
  if (more_in_flight) __builtin_amdgcn_s_wait_asynccnt(ACHUNK);
  else                __builtin_amdgcn_s_wait_asynccnt(0);
#  else
  if (more_in_flight) asm volatile("s_wait_asynccnt 27" ::: "memory");
  else                asm volatile("s_wait_asynccnt 0"  ::: "memory");
#  endif
#else
  (void)more_in_flight;
#endif
}

// ---- WMMA fragment loaders -------------------------------------------------
// A fragment (16x32 f16): lanes 0-15 K={0..7,16..23}, lanes 16-31 K={8..15,24..31}
__device__ __forceinline__ v16h load_a_frag(const _Float16* row, int ks, int lane) {
  const int kb = ks * 32 + ((lane & 31) < 16 ? 0 : 8);
  union { v16h h; uint4 u[2]; } f;
  f.u[0] = *(const uint4*)(row + kb);
  f.u[1] = *(const uint4*)(row + kb + 16);
  return f.h;
}

__device__ __forceinline__ v16h load_b_frag(const _Float16* p) {
  union { v16h h; uint4 u[2]; } f;
  f.u[0] = *(const uint4*)(p);
  f.u[1] = *(const uint4*)(p + 8);
  return f.h;
}

// 16(M) x 16(N per wave) x 128(K) GEMM tile; A rows all valid (16 = 2 batches x 8 tok)
__device__ __forceinline__ v8f gemm16(const _Float16* sArows, const _Float16* Wsw,
                                      int nt, int lane) {
  const _Float16* row = sArows + (size_t)(lane & 15) * SRL;
  v8f c = {};
#pragma unroll
  for (int ks = 0; ks < 4; ++ks) {
    v16h a = load_a_frag(row, ks, lane);
    v16h w = load_b_frag(Wsw + (((size_t)nt * 4 + ks) * 32 + lane) * 16);
    c = __builtin_amdgcn_wmma_f32_16x16x32_f16(false, a, false, w, (short)0, c,
                                               false, false);
  }
  return c;
}

// ---- prep: swizzle weights f32 -> f16 fragment order -----------------------
__global__ __launch_bounds__(256)
void prep_weights(const float* __restrict__ Wk, const float* __restrict__ Wv,
                  const float* __restrict__ Wp, const float* __restrict__ Wf1,
                  const float* __restrict__ Wf2, _Float16* __restrict__ swz) {
  const float* Ws[5] = {Wk, Wv, Wp, Wf1, Wf2};
  for (size_t idx = (size_t)blockIdx.x * 256 + threadIdx.x; idx < 5u * MATH;
       idx += (size_t)gridDim.x * 256) {
    int mat  = (int)(idx / MATH);
    int rem  = (int)(idx % MATH);
    int e    = rem & 15;
    int lane = (rem >> 4) & 31;
    int ks   = (rem >> 9) & 3;
    int nt   = (rem >> 11) & 7;
    int k = ks * 32 + ((lane < 16) ? 0 : 8) + (e < 8 ? e : e + 8);
    int n = nt * 16 + (lane & 15);
    swz[idx] = (_Float16)Ws[mat][k * QDIM + n];
  }
}

// ---- prep: q = LN(query) @ Wq + bq, pre-scaled by 0.25 ---------------------
__global__ __launch_bounds__(256)
void prep_q(const float* __restrict__ query, const float* __restrict__ g,
            const float* __restrict__ bvec, const float* __restrict__ Wq,
            const float* __restrict__ bq, float* __restrict__ qs) {
  __shared__ float sQ[8 * QDIM];
  __shared__ float sM[8], sR[8];
  const int tid = threadIdx.x;
  for (int i = tid; i < 8 * QDIM; i += 256) sQ[i] = query[i];
  __syncthreads();
  if (tid < 8) {
    float m = 0.f;
    for (int k = 0; k < QDIM; ++k) m += sQ[tid * QDIM + k];
    m *= (1.0f / QDIM);
    float v = 0.f;
    for (int k = 0; k < QDIM; ++k) { float d = sQ[tid * QDIM + k] - m; v += d * d; }
    v *= (1.0f / QDIM);
    sM[tid] = m;
    sR[tid] = rsqrtf(v + 1e-5f);
  }
  __syncthreads();
  for (int i = tid; i < 8 * QDIM; i += 256) {
    int tok = i >> 7, j = i & 127;
    float acc = bq[j];
    for (int k = 0; k < QDIM; ++k) {
      float qn = (sQ[tok * QDIM + k] - sM[tok]) * sR[tok] * g[k] + bvec[k];
      acc += qn * Wq[k * QDIM + j];
    }
    qs[i] = acc * 0.25f;
  }
}

// ---- main fused kernel: persistent block, PPB batch-pairs, double-buffered -
__global__ __launch_bounds__(256)
void fused_cross_attn(const float* __restrict__ X,      // [B,54,128]
                      const float* __restrict__ query,  // [8,128]
                      const float* __restrict__ qs,     // [8,128] scaled q
                      const _Float16* __restrict__ swz,
                      const float* __restrict__ bk, const float* __restrict__ bv,
                      const float* __restrict__ bp, const float* __restrict__ bf1,
                      const float* __restrict__ bf2,
                      const float* __restrict__ ln2g, const float* __restrict__ ln2b,
                      float* __restrict__ out) {
  extern __shared__ char smem[];
  float*    xraw[2];
  xraw[0]        = (float*)(smem + OFF_XRAW0);
  xraw[1]        = (float*)(smem + OFF_XRAW1);
  _Float16* sX   = (_Float16*)(smem + OFF_SX);
  _Float16* sK   = (_Float16*)(smem + OFF_SK);
  _Float16* sV   = (_Float16*)(smem + OFF_SV);
  float*    sAtt = (float*)(smem + OFF_ATT);
  float*    sXres= (float*)(smem + OFF_XRES);
  _Float16* sA   = (_Float16*)(smem + OFF_SA);
  _Float16* sB2  = (_Float16*)(smem + OFF_SB2);
  float*    sStat= (float*)(smem + OFF_STAT);   // [0..15]=mean, [16..31]=rstd

  const int tid  = threadIdx.x;
  const int wave = tid >> 5;
  const int lane = tid & 31;
  const int l16  = lane & 15;
  const int pair0 = blockIdx.x * PPB;

  // hoist this wave's projection B fragments (reused across all pairs)
  const int mat    = wave & 1;           // 0 = K, 1 = V
  const int ntBase = (wave >> 1) * 2;
  const _Float16* WswProj = swz + (size_t)mat * MATH;
  __builtin_prefetch(WswProj + ((size_t)ntBase * 4 * 32) * 16, 0, 0);
  v16h bfr[2][4];
#pragma unroll
  for (int t = 0; t < 2; ++t)
#pragma unroll
    for (int ks = 0; ks < 4; ++ks)
      bfr[t][ks] =
          load_b_frag(WswProj + (((size_t)(ntBase + t) * 4 + ks) * 32 + lane) * 16);

  // prologue: stage first pair into buffer 0
  stage_pair(X + (size_t)pair0 * PAIR_FLOATS, xraw[0], tid);

  for (int pi = 0; pi < PPB; ++pi) {
    const int pair = pair0 + pi;
    float* cur = xraw[pi & 1];
    if (pi + 1 < PPB)
      stage_pair(X + (size_t)(pair + 1) * PAIR_FLOATS, xraw[(pi + 1) & 1], tid);
    wait_stage(pi + 1 < PPB);
    __syncthreads();

    // 1) convert raw f32 pair tile -> f16 (rows 108..111 zero)
    for (int i = tid; i < MPAD * QDIM; i += 256) {
      int r = i >> 7, c = i & 127;
      sX[r * SXL + c] = (r < PROWS) ? (_Float16)cur[i] : (_Float16)0.f;
    }
    __syncthreads();

    // 2) K/V projection: [112x128] x [128x128] via WMMA
    for (int t = 0; t < 2; ++t) {
      const int nt = ntBase + t;
      const int n  = nt * 16 + l16;
      const float bb = (mat ? bv : bk)[n];
      _Float16* dst = mat ? sV : sK;
      for (int mt = 0; mt < MTILES; ++mt) {
        const _Float16* row = sX + (size_t)(mt * 16 + l16) * SXL;
        v8f c = {};
#pragma unroll
        for (int ks = 0; ks < 4; ++ks) {
          v16h a = load_a_frag(row, ks, lane);
          c = __builtin_amdgcn_wmma_f32_16x16x32_f16(false, a, false, bfr[t][ks],
                                                     (short)0, c, false, false);
        }
#pragma unroll
        for (int r = 0; r < 8; ++r) {
          int m = mt * 16 + r + ((lane < 16) ? 0 : 8);
          if (m < PROWS) dst[m * QDIM + n] = (_Float16)(c[r] + bb);
        }
      }
    }
    __syncthreads();

    // 3) regional attention: 128 (batch-in-pair, token, head) triples
    if (tid < 128) {
      const int bb2  = tid >> 6;              // batch within pair
      const int tok  = (tid >> 3) & 7;
      const int head = tid & 7;
      const int reg  = (tok < 2) ? tok : tok - 2;
      const int hb   = (reg / 3) * 3, wb = (reg % 3) * 3;
      const float* qv = qs + tok * QDIM + head * HDIM;
      const int rowOff = bb2 * LROWS;
      float logit[9];
      float mx = -1e30f;
#pragma unroll
      for (int i = 0; i < 9; ++i) {
        int l = (hb + i / 3) * 9 + (wb + i % 3);
        const _Float16* kp = sK + (rowOff + l) * QDIM + head * HDIM;
        float d = 0.f;
#pragma unroll
        for (int e = 0; e < HDIM; ++e) d += qv[e] * (float)kp[e];
        logit[i] = d;
        mx = fmaxf(mx, d);
      }
      float den = 0.f;
#pragma unroll
      for (int i = 0; i < 9; ++i) { logit[i] = __expf(logit[i] - mx); den += logit[i]; }
      const float inv = 1.0f / den;
      float acc[HDIM];
#pragma unroll
      for (int e = 0; e < HDIM; ++e) acc[e] = 0.f;
#pragma unroll
      for (int i = 0; i < 9; ++i) {
        int l = (hb + i / 3) * 9 + (wb + i % 3);
        const _Float16* vp = sV + (rowOff + l) * QDIM + head * HDIM;
        float w = logit[i] * inv;
#pragma unroll
        for (int e = 0; e < HDIM; ++e) acc[e] += w * (float)vp[e];
      }
#pragma unroll
      for (int e = 0; e < HDIM; ++e)
        sAtt[(bb2 * 8 + tok) * QDIM + head * HDIM + e] = acc[e];
    }
    __syncthreads();

    // 4) attention result -> f16 A tile (all 16 rows valid)
    for (int i = tid; i < 16 * QDIM; i += 256) {
      int m = i >> 7, k = i & 127;
      sA[m * SRL + k] = (_Float16)sAtt[i];
    }
    __syncthreads();

    // 5) x = query + att @ Wp + bp
    {
      v8f c = gemm16(sA, swz + 2 * (size_t)MATH, wave, lane);
      const int n = wave * 16 + l16;
      const float bb = bp[n];
#pragma unroll
      for (int r = 0; r < 8; ++r) {
        int m = r + ((lane < 16) ? 0 : 8);       // token = r for both batches
        sXres[m * QDIM + n] = query[r * QDIM + n] + c[r] + bb;
      }
    }
    __syncthreads();

    // 6) LayerNorm2 -> f16 A tile
    if (tid < 16) {
      float m = 0.f;
      for (int k = 0; k < QDIM; ++k) m += sXres[tid * QDIM + k];
      m *= (1.0f / QDIM);
      float v = 0.f;
      for (int k = 0; k < QDIM; ++k) { float d = sXres[tid * QDIM + k] - m; v += d * d; }
      v *= (1.0f / QDIM);
      sStat[tid]      = m;
      sStat[16 + tid] = rsqrtf(v + 1e-5f);
    }
    __syncthreads();
    for (int i = tid; i < 16 * QDIM; i += 256) {
      int m = i >> 7, k = i & 127;
      sA[m * SRL + k] =
          (_Float16)((sXres[i] - sStat[m]) * sStat[16 + m] * ln2g[k] + ln2b[k]);
    }
    __syncthreads();

    // 7) h = gelu_exact(LN(x) @ Wf1 + bf1)
    {
      v8f c = gemm16(sA, swz + 3 * (size_t)MATH, wave, lane);
      const int n = wave * 16 + l16;
      const float bb = bf1[n];
#pragma unroll
      for (int r = 0; r < 8; ++r) {
        int m = r + ((lane < 16) ? 0 : 8);
        float x = c[r] + bb;
        float hv = 0.5f * x * (1.0f + erff(x * 0.70710678118654752f));
        sB2[m * SRL + n] = (_Float16)hv;
      }
    }
    __syncthreads();

    // 8) out = x + h @ Wf2 + bf2
    {
      v8f c = gemm16(sB2, swz + 4 * (size_t)MATH, wave, lane);
      const int n = wave * 16 + l16;
      const float bb = bf2[n];
      float* ob = out + (size_t)pair * (16 * QDIM);
#pragma unroll
      for (int r = 0; r < 8; ++r) {
        int m = r + ((lane < 16) ? 0 : 8);
        ob[m * QDIM + n] = sXres[m * QDIM + n] + c[r] + bb;
      }
    }
    __syncthreads();
  }
}

extern "C" void kernel_launch(void* const* d_in, const int* in_sizes, int n_in,
                              void* d_out, int out_size, void* d_ws, size_t ws_size,
                              hipStream_t stream) {
  const float* query = (const float*)d_in[0];
  const float* tgt   = (const float*)d_in[1];
  const float* ln1g  = (const float*)d_in[2];
  const float* ln1b  = (const float*)d_in[3];
  const float* ln2g  = (const float*)d_in[4];
  const float* ln2b  = (const float*)d_in[5];
  const float* Wq    = (const float*)d_in[6];
  const float* bq    = (const float*)d_in[7];
  const float* Wk    = (const float*)d_in[8];
  const float* bk    = (const float*)d_in[9];
  const float* Wv    = (const float*)d_in[10];
  const float* bv    = (const float*)d_in[11];
  const float* Wp    = (const float*)d_in[12];
  const float* bp    = (const float*)d_in[13];
  const float* Wf1   = (const float*)d_in[14];
  const float* bf1   = (const float*)d_in[15];
  const float* Wf2   = (const float*)d_in[16];
  const float* bf2   = (const float*)d_in[17];
  float* out = (float*)d_out;

  float*    qs  = (float*)d_ws;
  _Float16* swz = (_Float16*)((char*)d_ws + 4096);

  const int B      = in_sizes[1] / (LROWS * QDIM);  // 8192
  const int pairs  = B / PAIRB;                     // 4096
  const int blocks = pairs / PPB;                   // 1024

  (void)hipFuncSetAttribute((const void*)fused_cross_attn,
                            hipFuncAttributeMaxDynamicSharedMemorySize, SMEM_BYTES);

  prep_weights<<<80, 256, 0, stream>>>(Wk, Wv, Wp, Wf1, Wf2, swz);
  prep_q<<<1, 256, 0, stream>>>(query, ln1g, ln1b, Wq, bq, qs);
  fused_cross_attn<<<blocks, 256, SMEM_BYTES, stream>>>(
      tgt, query, qs, swz, bk, bv, bp, bf1, bf2, ln2g, ln2b, out);
}
